// TinyTransformer_23622320128268
// MI455X (gfx1250) — compile-verified, matrix operand order checked by reading, not run on hardware
//
#include <hip/hip_runtime.h>
#include <hip/hip_bf16.h>

// ---------------------------------------------------------------------------
// Types for CDNA5 WMMA (wave32): v_wmma_f32_16x16x32_bf16
// ---------------------------------------------------------------------------
typedef __attribute__((ext_vector_type(16))) __bf16 v16bf;
typedef __attribute__((ext_vector_type(8)))  __bf16 v8bf;
typedef __attribute__((ext_vector_type(8)))  float  v8f;

#define WAVES_PER_BLOCK 8   // 256 threads, wave32

enum { EPI_NONE = 0, EPI_BIAS = 1, EPI_BIAS_GELU = 2 };
enum { OUT_F32 = 0, OUT_BF16 = 1, OUT_BF16_TR = 2 };

// ---------------------------------------------------------------------------
// Fragment loaders (per ISA 7.12.2 layouts, wave32)
// A 16x32 bf16: lane&15 = M row; lanes<16 hold K = k0+[0..7] and k0+16+[0..7];
// lanes>=16 hold K = k0+8+[0..7] and k0+24+[0..7].
// B 32x16 bf16 from W[N,K] row-major is the mirror layout (lane&15 = N col),
// so both fragments are two contiguous 16-byte loads per lane.
// ---------------------------------------------------------------------------
__device__ inline v16bf frag_combine(v8bf lo, v8bf hi) {
    v16bf r;
#pragma unroll
    for (int i = 0; i < 8; ++i) { r[i] = lo[i]; r[i + 8] = hi[i]; }
    return r;
}

__device__ inline v16bf frag_zero() {
    v16bf r;
#pragma unroll
    for (int i = 0; i < 16; ++i) r[i] = (__bf16)0.0f;
    return r;
}

__device__ inline v16bf load_frag_a(const __bf16* __restrict__ A, int lda,
                                    int m0, int k0, int lane) {
    int m  = m0 + (lane & 15);
    int kb = k0 + ((lane >> 4) << 3);
    const __bf16* p = A + (long)m * lda + kb;
    v8bf lo = *reinterpret_cast<const v8bf*>(p);
    v8bf hi = *reinterpret_cast<const v8bf*>(p + 16);
    return frag_combine(lo, hi);
}

__device__ inline v16bf load_frag_b(const __bf16* __restrict__ W, int ldw,
                                    int n0, int k0, int lane) {
    int n  = n0 + (lane & 15);
    int kb = k0 + ((lane >> 4) << 3);
    const __bf16* p = W + (long)n * ldw + kb;
    v8bf lo = *reinterpret_cast<const v8bf*>(p);
    v8bf hi = *reinterpret_cast<const v8bf*>(p + 16);
    return frag_combine(lo, hi);
}

__device__ inline v16bf load_frag_b_guard(const __bf16* __restrict__ W, int ldw,
                                          int n0, int k0, int Ndim, int lane) {
    int n = n0 + (lane & 15);
    if (n >= Ndim) return frag_zero();
    int kb = k0 + ((lane >> 4) << 3);
    const __bf16* p = W + (long)n * ldw + kb;
    v8bf lo = *reinterpret_cast<const v8bf*>(p);
    v8bf hi = *reinterpret_cast<const v8bf*>(p + 16);
    return frag_combine(lo, hi);
}

// ---------------------------------------------------------------------------
// Generic WMMA GEMM:  Out = epilogue( scale * A[M,K] @ W^T + bias ) (+resid)
//   A: bf16 row-major [M,K] (lda);  W: bf16 row-major [N,K] (ldb)
// Each wave computes a (16*MT) x (16*NT) tile: B frags reused MT times,
// A frags reused NT times. blockIdx.y = batch (strides sA/sB/sO/sR, elems).
// OUT_BF16_TR stores out[n*ldo + m] (per-batch transposed output).
// ---------------------------------------------------------------------------
template <int MT, int NT, int NGUARD, int EPI, int OUTT, int CAUSAL, int RESID>
__global__ __launch_bounds__(256) void gemm_wmma_kernel(
    const __bf16* __restrict__ A, const __bf16* __restrict__ W,
    const float* __restrict__ bias, const float* __restrict__ resid,
    void* __restrict__ outp,
    int Mdim, int Ndim, int Kdim, int lda, int ldb, int ldo,
    long sA, long sB, long sO, long sR, float scale)
{
    const int lane = threadIdx.x & 31;
    const int wave = threadIdx.x >> 5;
    const long tilesN = (Ndim + 16 * NT - 1) / (16 * NT);
    const long tilesM = (Mdim + 16 * MT - 1) / (16 * MT);
    long tile = (long)blockIdx.x * WAVES_PER_BLOCK + wave;
    if (tile >= tilesM * tilesN) return;
    const int tm = (int)(tile / tilesN);
    const int tn = (int)(tile % tilesN);
    const int m0 = tm * 16 * MT;
    const int n0 = tn * 16 * NT;
    if (CAUSAL && n0 > m0 + 16 * MT - 1) return;  // strictly above the diagonal

    const int b = blockIdx.y;
    const __bf16* Ab = A + (long)b * sA;
    const __bf16* Wb = W + (long)b * sB;

    const v8f vz = {0.f, 0.f, 0.f, 0.f, 0.f, 0.f, 0.f, 0.f};
    v8f acc[MT][NT];
#pragma unroll
    for (int mi = 0; mi < MT; ++mi)
#pragma unroll
        for (int t = 0; t < NT; ++t) acc[mi][t] = vz;

    const __bf16* aprefetch = Ab + (long)(m0 + (lane & 15)) * lda;

    for (int k0 = 0; k0 < Kdim; k0 += 32) {
        __builtin_prefetch(aprefetch + k0 + 64, 0, 3);  // speculative; OOB dropped
        v16bf afr[MT];
#pragma unroll
        for (int mi = 0; mi < MT; ++mi)
            afr[mi] = load_frag_a(Ab, lda, m0 + 16 * mi, k0, lane);
#pragma unroll
        for (int t = 0; t < NT; ++t) {
            v16bf bfr = NGUARD
                ? load_frag_b_guard(Wb, ldb, n0 + t * 16, k0, Ndim, lane)
                : load_frag_b(Wb, ldb, n0 + t * 16, k0, lane);
#pragma unroll
            for (int mi = 0; mi < MT; ++mi)
                acc[mi][t] = __builtin_amdgcn_wmma_f32_16x16x32_bf16(
                    false, afr[mi], false, bfr, (short)0, acc[mi][t], false, false);
        }
    }

    // D layout: VGPR r, lanes0-15 -> M=m0+r, lanes16-31 -> M=m0+8+r; N = lane&15.
    const int nb = (lane & 15);
#pragma unroll
    for (int mi = 0; mi < MT; ++mi) {
        const int mb = m0 + 16 * mi + ((lane >> 4) << 3);
#pragma unroll
        for (int t = 0; t < NT; ++t) {
            const int n = n0 + t * 16 + nb;
            if (n >= Ndim) continue;
            const float bv = (EPI != EPI_NONE) ? bias[n] : 0.0f;
#pragma unroll
            for (int r = 0; r < 8; ++r) {
                const int m = mb + r;
                if (m >= Mdim) continue;
                float v = acc[mi][t][r] * scale + bv;
                if (EPI == EPI_BIAS_GELU)
                    v = 0.5f * v * (1.0f + erff(v * 0.70710678118654752f));
                if (RESID) v += resid[(long)b * sR + (long)m * ldo + n];
                if (OUTT == OUT_BF16_TR) {
                    ((__bf16*)outp)[(long)b * sO + (long)n * ldo + m] = (__bf16)v;
                } else {
                    const long oi = (long)b * sO + (long)m * ldo + n;
                    if (OUTT == OUT_BF16) ((__bf16*)outp)[oi] = (__bf16)v;
                    else                  ((float*)outp)[oi]  = v;
                }
            }
        }
    }
}

// ---------------------------------------------------------------------------
// Small helper kernels
// ---------------------------------------------------------------------------
__global__ void cvt_f32_to_bf16_kernel(const float* __restrict__ in,
                                       __bf16* __restrict__ out, long n) {
    long i = (long)blockIdx.x * blockDim.x + threadIdx.x;
    if (i < n) out[i] = (__bf16)in[i];
}

__global__ void embed_kernel(const int* __restrict__ x,
                             const float* __restrict__ emb,
                             float* __restrict__ h, long ntok, int D) {
    long i = (long)blockIdx.x * blockDim.x + threadIdx.x;
    if (i >= ntok * (long)D) return;
    long t = i / D;
    int  d = (int)(i - t * D);
    h[i] = emb[(long)x[t] * D + d];
}

__global__ __launch_bounds__(256) void layernorm_bf16_kernel(
    const float* __restrict__ in, const float* __restrict__ w,
    const float* __restrict__ b, __bf16* __restrict__ out, int D)
{
    const long row = blockIdx.x;
    const float* r = in + row * (long)D;
    __shared__ float s1[256], s2[256];
    float a = 0.f, a2 = 0.f;
    for (int j = threadIdx.x; j < D; j += 256) { float v = r[j]; a += v; a2 += v * v; }
    s1[threadIdx.x] = a; s2[threadIdx.x] = a2;
    __syncthreads();
    for (int st = 128; st > 0; st >>= 1) {
        if ((int)threadIdx.x < st) {
            s1[threadIdx.x] += s1[threadIdx.x + st];
            s2[threadIdx.x] += s2[threadIdx.x + st];
        }
        __syncthreads();
    }
    const float mean = s1[0] / D;
    const float var  = s2[0] / D - mean * mean;
    const float inv  = rsqrtf(var + 1e-5f);
    for (int j = threadIdx.x; j < D; j += 256) {
        float v = (r[j] - mean) * inv * w[j] + b[j];
        out[row * (long)D + j] = (__bf16)v;
    }
}

__global__ __launch_bounds__(256) void softmax_causal_kernel(
    const float* __restrict__ sc, __bf16* __restrict__ pr, int S)
{
    const int i = blockIdx.x;   // query row
    const int b = blockIdx.y;   // batch
    const float* row  = sc + ((long)b * S + i) * (long)S;
    __bf16*      orow = pr + ((long)b * S + i) * (long)S;
    const int len = i + 1;      // causal: attend to 0..i
    __shared__ float red[256];

    float mx = -3.4e38f;
    for (int j = threadIdx.x; j < len; j += 256) mx = fmaxf(mx, row[j]);
    red[threadIdx.x] = mx;
    __syncthreads();
    for (int st = 128; st > 0; st >>= 1) {
        if ((int)threadIdx.x < st)
            red[threadIdx.x] = fmaxf(red[threadIdx.x], red[threadIdx.x + st]);
        __syncthreads();
    }
    mx = red[0];
    __syncthreads();

    float sm = 0.f;
    for (int j = threadIdx.x; j < len; j += 256) sm += expf(row[j] - mx);
    red[threadIdx.x] = sm;
    __syncthreads();
    for (int st = 128; st > 0; st >>= 1) {
        if ((int)threadIdx.x < st) red[threadIdx.x] += red[threadIdx.x + st];
        __syncthreads();
    }
    const float inv = 1.0f / red[0];

    for (int j = threadIdx.x; j < len; j += 256)
        orow[j] = (__bf16)(expf(row[j] - mx) * inv);
    for (int j = len + threadIdx.x; j < S; j += 256)
        orow[j] = (__bf16)0.0f;
}

// ---------------------------------------------------------------------------
// Host launcher
// ---------------------------------------------------------------------------
static inline int gemm_blocks(long Mdim, long Ndim, int MT, int NT) {
    long tilesM = (Mdim + 16L * MT - 1) / (16L * MT);
    long tilesN = (Ndim + 16L * NT - 1) / (16L * NT);
    long tiles  = tilesM * tilesN;
    return (int)((tiles + WAVES_PER_BLOCK - 1) / WAVES_PER_BLOCK);
}

extern "C" void kernel_launch(void* const* d_in, const int* in_sizes, int n_in,
                              void* d_out, int out_size, void* d_ws, size_t ws_size,
                              hipStream_t stream)
{
    (void)in_sizes; (void)n_in; (void)out_size; (void)ws_size;
    const int   Bsz = 8, S = 2048, D = 1024, FF = 4096, V = 14;
    const long  M   = (long)Bsz * S;              // 16384 rows
    const float ATT_SCALE = 0.03125f;             // 1/sqrt(1024)

    const int*   x    = (const int*)  d_in[0];
    const float* temb = (const float*)d_in[1];
    const float* ln1w = (const float*)d_in[2];
    const float* ln1b = (const float*)d_in[3];
    const float* qw = (const float*)d_in[4];  const float* qb = (const float*)d_in[5];
    const float* kw = (const float*)d_in[6];  const float* kb = (const float*)d_in[7];
    const float* vw = (const float*)d_in[8];  const float* vb = (const float*)d_in[9];
    const float* pw = (const float*)d_in[10]; const float* pb = (const float*)d_in[11];
    const float* ln2w = (const float*)d_in[12];
    const float* ln2b = (const float*)d_in[13];
    const float* f1w = (const float*)d_in[14]; const float* f1b = (const float*)d_in[15];
    const float* f2w = (const float*)d_in[16]; const float* f2b = (const float*)d_in[17];
    const float* lnfw = (const float*)d_in[18];
    const float* lnfb = (const float*)d_in[19];
    float* logits = (float*)d_out;

    // ---- workspace bump allocator (256B aligned) ----
    char*  base = (char*)d_ws;
    size_t off  = 0;
    auto alloc = [&](size_t bytes) -> void* {
        off = (off + 255) & ~(size_t)255;
        void* p = base + off;
        off += bytes;
        return p;
    };
    __bf16* qw_b  = (__bf16*)alloc((size_t)D * D * 2);
    __bf16* kw_b  = (__bf16*)alloc((size_t)D * D * 2);
    __bf16* vw_b  = (__bf16*)alloc((size_t)D * D * 2);
    __bf16* pw_b  = (__bf16*)alloc((size_t)D * D * 2);
    __bf16* f1w_b = (__bf16*)alloc((size_t)FF * D * 2);
    __bf16* f2w_b = (__bf16*)alloc((size_t)D * FF * 2);
    __bf16* te_b  = (__bf16*)alloc((size_t)V * D * 2);
    float*  h     = (float*) alloc((size_t)M * D * 4);   // residual 1 / final h3
    float*  h2    = (float*) alloc((size_t)M * D * 4);   // residual 2
    __bf16* hn    = (__bf16*)alloc((size_t)M * D * 2);   // LN outputs (reused 3x)
    __bf16* qbuf  = (__bf16*)alloc((size_t)M * D * 2);   // also reused as attout
    __bf16* kbuf  = (__bf16*)alloc((size_t)M * D * 2);
    __bf16* vT    = (__bf16*)alloc((size_t)M * D * 2);   // V written transposed [b][D][S]
    float*  scores= (float*) alloc((size_t)Bsz * S * S * 4); // reused as ff1 (bf16)
    __bf16* probs = (__bf16*)alloc((size_t)Bsz * S * S * 2);
    __bf16* ff1   = (__bf16*)scores;
    __bf16* attout= qbuf;
    float*  h3    = h;

    // ---- 1) convert weights fp32 -> bf16 (fits in 192MB L2 thereafter) ----
    auto cvt = [&](const float* in, __bf16* out, long n) {
        cvt_f32_to_bf16_kernel<<<(int)((n + 255) / 256), 256, 0, stream>>>(in, out, n);
    };
    cvt(qw,  qw_b,  (long)D * D);
    cvt(kw,  kw_b,  (long)D * D);
    cvt(vw,  vw_b,  (long)D * D);
    cvt(pw,  pw_b,  (long)D * D);
    cvt(f1w, f1w_b, (long)FF * D);
    cvt(f2w, f2w_b, (long)D * FF);
    cvt(temb, te_b, (long)V * D);

    // ---- 2) embedding gather ----
    embed_kernel<<<(int)((M * D + 255) / 256), 256, 0, stream>>>(x, temb, h, M, D);

    // ---- 3) LN1 -> hn (bf16) ----
    layernorm_bf16_kernel<<<(int)M, 256, 0, stream>>>(h, ln1w, ln1b, hn, D);

    // ---- 4) Q, K projections (row-major out); V projection (transposed out) ----
    {
        dim3 g(gemm_blocks(M, D, 2, 4), 1);
        gemm_wmma_kernel<2, 4, 0, EPI_BIAS, OUT_BF16, 0, 0><<<g, 256, 0, stream>>>(
            hn, qw_b, qb, nullptr, qbuf, (int)M, D, D, D, D, D, 0, 0, 0, 0, 1.0f);
        gemm_wmma_kernel<2, 4, 0, EPI_BIAS, OUT_BF16, 0, 0><<<g, 256, 0, stream>>>(
            hn, kw_b, kb, nullptr, kbuf, (int)M, D, D, D, D, D, 0, 0, 0, 0, 1.0f);
    }
    {   // batched so each batch's V lands as vT[b] = [D, S] row-major
        dim3 g(gemm_blocks(S, D, 2, 4), Bsz);
        gemm_wmma_kernel<2, 4, 0, EPI_BIAS, OUT_BF16_TR, 0, 0><<<g, 256, 0, stream>>>(
            hn, vw_b, vb, nullptr, vT, S, D, D, D, D, S,
            (long)S * D, 0, (long)D * S, 0, 1.0f);
    }

    // ---- 5) scores = scale * q @ k^T, lower-triangle 32x32 tiles only ----
    {
        dim3 g(gemm_blocks(S, S, 2, 2), Bsz);
        gemm_wmma_kernel<2, 2, 0, EPI_NONE, OUT_F32, 1, 0><<<g, 256, 0, stream>>>(
            qbuf, kbuf, nullptr, nullptr, scores, S, S, D, D, D, S,
            (long)S * D, (long)S * D, (long)S * S, 0, ATT_SCALE);
    }

    // ---- 6) causal softmax -> probs (bf16, zero above diagonal) ----
    {
        dim3 g(S, Bsz);
        softmax_causal_kernel<<<g, 256, 0, stream>>>(scores, probs, S);
    }

    // ---- 7) attout = probs @ v  via vT as W[N=D, K=S] (contiguous frags) ----
    {
        dim3 g(gemm_blocks(S, D, 2, 4), Bsz);
        gemm_wmma_kernel<2, 4, 0, EPI_NONE, OUT_BF16, 0, 0><<<g, 256, 0, stream>>>(
            probs, vT, nullptr, nullptr, attout, S, D, S, S, S, D,
            (long)S * S, (long)D * S, (long)S * D, 0, 1.0f);
    }

    // ---- 8) h2 = attout @ pw^T + pb + h (residual) -> fp32 ----
    {
        dim3 g(gemm_blocks(M, D, 2, 4), 1);
        gemm_wmma_kernel<2, 4, 0, EPI_BIAS, OUT_F32, 0, 1><<<g, 256, 0, stream>>>(
            attout, pw_b, pb, h, h2, (int)M, D, D, D, D, D, 0, 0, 0, 0, 1.0f);
    }

    // ---- 9) LN2 -> hn ----
    layernorm_bf16_kernel<<<(int)M, 256, 0, stream>>>(h2, ln2w, ln2b, hn, D);

    // ---- 10) ff1 = gelu(hn @ f1w^T + f1b) -> bf16 [M,4096] ----
    {
        dim3 g(gemm_blocks(M, FF, 2, 4), 1);
        gemm_wmma_kernel<2, 4, 0, EPI_BIAS_GELU, OUT_BF16, 0, 0><<<g, 256, 0, stream>>>(
            hn, f1w_b, f1b, nullptr, ff1, (int)M, FF, D, D, D, FF, 0, 0, 0, 0, 1.0f);
    }

    // ---- 11) h3 = ff1 @ f2w^T + f2b + h2 (residual) -> fp32 ----
    {
        dim3 g(gemm_blocks(M, D, 2, 4), 1);
        gemm_wmma_kernel<2, 4, 0, EPI_BIAS, OUT_F32, 0, 1><<<g, 256, 0, stream>>>(
            ff1, f2w_b, f2b, h2, h3, (int)M, D, FF, FF, FF, D, 0, 0, 0, 0, 1.0f);
    }

    // ---- 12) final LN -> hn ----
    layernorm_bf16_kernel<<<(int)M, 256, 0, stream>>>(h3, lnfw, lnfb, hn, D);

    // ---- 13) logits = hn @ tok_emb^T (N=14, B-frag + store guarded) ----
    {
        dim3 g(gemm_blocks(M, V, 2, 1), 1);
        gemm_wmma_kernel<2, 1, 1, EPI_NONE, OUT_F32, 0, 0><<<g, 256, 0, stream>>>(
            hn, te_b, nullptr, nullptr, logits, (int)M, V, D, D, D, V, 0, 0, 0, 0, 1.0f);
    }
}
